// QNet_14499809592003
// MI455X (gfx1250) — matrix operand order repaired and checked by reading, not compiled, not run on hardware
//
#include <hip/hip_runtime.h>

#define N_NODES 50000
#define N_EDGES 800000
#define C 64

typedef __attribute__((ext_vector_type(16))) _Float16 v16h;
typedef __attribute__((ext_vector_type(8)))  float    v8f;

union Frag16 {
    v16h v;
    uint4 q[2];
};

// LDS byte-offset of a __shared__ object (addrspace(3) pointers are 32-bit
// offsets from the wave's LDS base, which is what the async-LDS ops consume).
__device__ __forceinline__ unsigned lds_off_u32(const void* p) {
    return (unsigned)(unsigned long long)(__attribute__((address_space(3))) const void*)p;
}

// ---------------------------------------------------------------------------
// K0: one-time pack of me_W1 (256x128 f32) into f16 in workspace, so the hot
// kernel can stage it into LDS with GLOBAL_LOAD_ASYNC_TO_LDS_B128 (raw bytes).
// ---------------------------------------------------------------------------
__global__ void pack_w1_half(const float* __restrict__ w, _Float16* __restrict__ wh) {
    int i = blockIdx.x * blockDim.x + threadIdx.x;
    if (i < 256 * 128) wh[i] = (_Float16)w[i];
}

// ---------------------------------------------------------------------------
// K1: LEConv1 linears (F_IN=4 -> C=64). One thread per (node, channel).
// ---------------------------------------------------------------------------
__global__ void conv1_lin(const float* __restrict__ x,
                          const float* __restrict__ W1, const float* __restrict__ b1,
                          const float* __restrict__ W2,
                          const float* __restrict__ W3, const float* __restrict__ b3,
                          float* __restrict__ aBuf, float* __restrict__ bBuf,
                          float* __restrict__ qBuf) {
    int tid = blockIdx.x * blockDim.x + threadIdx.x;
    if (tid >= N_NODES * C) return;
    int i = tid >> 6;
    int c = tid & 63;
    float4 xv = *(const float4*)(x + i * 4);
    float4 w1 = *(const float4*)(W1 + c * 4);
    float4 w2 = *(const float4*)(W2 + c * 4);
    float4 w3 = *(const float4*)(W3 + c * 4);
    float a = fmaf(xv.x, w1.x, fmaf(xv.y, w1.y, fmaf(xv.z, w1.z, xv.w * w1.w))) + b1[c];
    float b = fmaf(xv.x, w2.x, fmaf(xv.y, w2.y, fmaf(xv.z, w2.z, xv.w * w2.w)));
    float q = fmaf(xv.x, w3.x, fmaf(xv.y, w3.y, fmaf(xv.z, w3.z, xv.w * w3.w))) + b3[c];
    aBuf[tid] = a;
    bBuf[tid] = b;
    qBuf[tid] = q;
}

// ---------------------------------------------------------------------------
// K2: edge scatter: q[dst] += ew * (a[src] - b[dst]).
// a/b/q are 12.8 MB each -> L2 resident; L2-atomic bound by design.
// ---------------------------------------------------------------------------
__global__ void edge_scatter(const int* __restrict__ src, const int* __restrict__ dst,
                             const float* __restrict__ ew,
                             const float* __restrict__ aBuf, const float* __restrict__ bBuf,
                             float* __restrict__ qBuf) {
    int tid = blockIdx.x * blockDim.x + threadIdx.x;
    if (tid >= N_EDGES * 16) return;
    int e  = tid >> 4;
    int c0 = (tid & 15) * 4;
    int s = src[e];
    int d = dst[e];
    float w = ew[e];
    float4 av = *(const float4*)(aBuf + (size_t)s * C + c0);
    float4 bv = *(const float4*)(bBuf + (size_t)d * C + c0);
    float* qp = qBuf + (size_t)d * C + c0;
    atomicAdd(qp + 0, w * (av.x - bv.x));
    atomicAdd(qp + 1, w * (av.y - bv.y));
    atomicAdd(qp + 2, w * (av.z - bv.z));
    atomicAdd(qp + 3, w * (av.w - bv.w));
}

// ---------------------------------------------------------------------------
// K3: LEConv2 linears (64 -> 64). One block per node, row staged in LDS.
// ---------------------------------------------------------------------------
__global__ void conv2_lin(const float* __restrict__ qin,
                          const float* __restrict__ W1, const float* __restrict__ b1,
                          const float* __restrict__ W2,
                          const float* __restrict__ W3, const float* __restrict__ b3,
                          float* __restrict__ aBuf, float* __restrict__ bBuf,
                          float* __restrict__ qBuf) {
    __shared__ float row[C];
    int i = blockIdx.x;
    int c = threadIdx.x;
    row[c] = qin[(size_t)i * C + c];
    __syncthreads();
    const float* w1 = W1 + c * C;
    const float* w2 = W2 + c * C;
    const float* w3 = W3 + c * C;
    float a = b1[c];
    float b = 0.0f;
    float q = b3[c];
#pragma unroll 8
    for (int k = 0; k < C; ++k) {
        float r = row[k];
        a = fmaf(r, w1[k], a);
        b = fmaf(r, w2[k], b);
        q = fmaf(r, w3[k], q);
    }
    aBuf[(size_t)i * C + c] = a;
    bBuf[(size_t)i * C + c] = b;
    qBuf[(size_t)i * C + c] = q;
}

// ---------------------------------------------------------------------------
// K5: fused edge MLP via WMMA.
// me_W1 (f16, pre-packed) staged into LDS with GLOBAL_LOAD_ASYNC_TO_LDS_B128
// (ASYNCcnt path). Per wave: 16 edges; h = relu(edge_rep @ W1.T + b1) as
// 16 N-tiles x 4 K-chunks of v_wmma_f32_16x16x32_f16 with inline-0 C on the
// first WMMA of each tile (bias folded into the ReLU epilogue); 256->4 layer
// fused in VALU (co-executes with the TRANS-class WMMAs).
// ---------------------------------------------------------------------------
__global__ __launch_bounds__(256) void edge_mlp(
        const float* __restrict__ qn,
        const _Float16* __restrict__ w1h,
        const int* __restrict__ src, const int* __restrict__ dst,
        const float* __restrict__ meB1,
        const float* __restrict__ meW2, const float* __restrict__ meB2,
        float* __restrict__ out) {
    __shared__ __align__(16) _Float16 sW1[256 * 128];      // 64 KB, [n][k]
    __shared__ __align__(16) _Float16 sEdge[8][16 * 128];  // 32 KB, per-wave [m][k]
    __shared__ float sB1[256];
    __shared__ float sW2[4 * 256];
    __shared__ float sB2[4];

    int t = threadIdx.x;

    // Stage W1 via async global->LDS copies: 4096 x 16B, 16 issues/thread.
    for (int idx = t; idx < 4096; idx += 256) {
        unsigned lo = lds_off_u32(&sW1[idx * 8]);
        const _Float16* gp = w1h + (size_t)idx * 8;
        asm volatile("global_load_async_to_lds_b128 %0, %1, off"
                     :: "v"(lo), "v"(gp) : "memory");
    }
    sB1[t] = meB1[t];
    for (int idx = t; idx < 1024; idx += 256) sW2[idx] = meW2[idx];
    if (t < 4) sB2[t] = meB2[t];
    asm volatile("s_wait_asynccnt 0x0" ::: "memory");
    __syncthreads();

    const int wave = t >> 5;
    const int lane = t & 31;
    const int e0 = (blockIdx.x * 8 + wave) * 16;   // exact: 6250*8*16 = 800000

    // Gather q[src]|q[dst] for 16 edges into f16 LDS tile.
    _Float16* ed = &sEdge[wave][0];
    const int g  = lane >> 4;
    const int cl = (lane & 15) * 4;
    for (int r = 0; r < 16; ++r) {
        int e = e0 + r;
        int node = g ? dst[e] : src[e];
        float4 v = *(const float4*)(qn + (size_t)node * C + cl);
        int o = r * 128 + g * 64 + cl;
        ed[o + 0] = (_Float16)v.x;
        ed[o + 1] = (_Float16)v.y;
        ed[o + 2] = (_Float16)v.z;
        ed[o + 3] = (_Float16)v.w;
    }
    __syncthreads();

    // A fragments: ISA 16-bit A 16x32 layout: lane L -> row M=L%16,
    // halves 0-7 = K 8g..8g+7, halves 8-15 = K 16+8g..16+8g+7 (g = L>>4)
    Frag16 A[4];
    {
        const int r  = lane & 15;
        const int kg = (lane >> 4) * 8;
#pragma unroll
        for (int kc = 0; kc < 4; ++kc) {
            const _Float16* p = ed + r * 128 + kc * 32 + kg;
            A[kc].q[0] = *(const uint4*)(p);
            A[kc].q[1] = *(const uint4*)(p + 16);
        }
    }

    const int nlane = lane & 15;
    const int bkg   = (lane >> 4) * 8;    // B layout: K 8g..8g+7, 16+8g..16+8g+7
    float acc2[8][4];
#pragma unroll
    for (int v = 0; v < 8; ++v)
#pragma unroll
        for (int j = 0; j < 4; ++j) acc2[v][j] = 0.0f;

#pragma unroll
    for (int nt = 0; nt < 16; ++nt) {
        const int col = nt * 16 + nlane;
        v8f cacc = {};   // lowers to inline-0 SRC2 on the first WMMA

#pragma unroll
        for (int kc = 0; kc < 4; ++kc) {
            Frag16 B;
            const _Float16* p = &sW1[col * 128 + kc * 32 + bkg];
            B.q[0] = *(const uint4*)(p);
            B.q[1] = *(const uint4*)(p + 16);
            cacc = __builtin_amdgcn_wmma_f32_16x16x32_f16(
                false, A[kc].v, false, B.v, (short)0, cacc, false, false);
        }

        // bias + ReLU + fused second layer partials. C/D layout: VGPR v holds
        // M = v + 8*(lane>>4), N = col.
        const float bias = sB1[col];
#pragma unroll
        for (int v = 0; v < 8; ++v) {
            float hv = fmaxf(cacc[v] + bias, 0.0f);
#pragma unroll
            for (int j = 0; j < 4; ++j)
                acc2[v][j] = fmaf(hv, sW2[j * 256 + col], acc2[v][j]);
        }
    }

    // Reduce over the 16 lanes sharing the same M-group.
#pragma unroll
    for (int off = 1; off < 16; off <<= 1)
#pragma unroll
        for (int v = 0; v < 8; ++v)
#pragma unroll
            for (int j = 0; j < 4; ++j)
                acc2[v][j] += __shfl_xor(acc2[v][j], off, 32);

    if ((lane & 15) < 4) {
        const int j = lane & 3;
        const int mbase = (lane >> 4) * 8;
#pragma unroll
        for (int v = 0; v < 8; ++v)
            out[(size_t)(e0 + mbase + v) * 4 + j] = acc2[v][j] + sB2[j];
    }
}

// ---------------------------------------------------------------------------
extern "C" void kernel_launch(void* const* d_in, const int* in_sizes, int n_in,
                              void* d_out, int out_size, void* d_ws, size_t ws_size,
                              hipStream_t stream) {
    const float* x    = (const float*)d_in[0];
    const int*   eidx = (const int*)d_in[1];
    const float* ew   = (const float*)d_in[2];
    const float* c1W1 = (const float*)d_in[3];
    const float* c1b1 = (const float*)d_in[4];
    const float* c1W2 = (const float*)d_in[5];
    const float* c1W3 = (const float*)d_in[6];
    const float* c1b3 = (const float*)d_in[7];
    const float* c2W1 = (const float*)d_in[8];
    const float* c2b1 = (const float*)d_in[9];
    const float* c2W2 = (const float*)d_in[10];
    const float* c2W3 = (const float*)d_in[11];
    const float* c2b3 = (const float*)d_in[12];
    const float* meW1 = (const float*)d_in[13];
    const float* meB1 = (const float*)d_in[14];
    const float* meW2 = (const float*)d_in[15];
    const float* meB2 = (const float*)d_in[16];
    const int* src = eidx;
    const int* dst = eidx + N_EDGES;

    float* ws   = (float*)d_ws;
    float* aBuf = ws;
    float* bBuf = aBuf + (size_t)N_NODES * C;
    float* q1   = bBuf + (size_t)N_NODES * C;
    float* q2   = q1   + (size_t)N_NODES * C;
    _Float16* w1h = (_Float16*)(q2 + (size_t)N_NODES * C);   // 64 KB f16 weights

    pack_w1_half<<<(256 * 128 + 255) / 256, 256, 0, stream>>>(meW1, w1h);
    conv1_lin<<<(N_NODES * C + 255) / 256, 256, 0, stream>>>(
        x, c1W1, c1b1, c1W2, c1W3, c1b3, aBuf, bBuf, q1);
    edge_scatter<<<(N_EDGES * 16 + 255) / 256, 256, 0, stream>>>(
        src, dst, ew, aBuf, bBuf, q1);
    conv2_lin<<<N_NODES, C, 0, stream>>>(
        q1, c2W1, c2b1, c2W2, c2W3, c2b3, aBuf, bBuf, q2);
    edge_scatter<<<(N_EDGES * 16 + 255) / 256, 256, 0, stream>>>(
        src, dst, ew, aBuf, bBuf, q2);
    edge_mlp<<<N_EDGES / (16 * 8), 256, 0, stream>>>(
        q2, w1h, src, dst, meB1, meW2, meB2, (float*)d_out);
}
